// MultiHeadSelfAttentionLayer_5188320494157
// MI455X (gfx1250) — compile-verified, matrix operand order checked by reading, not run on hardware
//
#include <hip/hip_runtime.h>
#include <hip/hip_bf16.h>

typedef __attribute__((ext_vector_type(16))) _Float16 v16h;
typedef __attribute__((ext_vector_type(8)))  _Float16 v8h;
typedef __attribute__((ext_vector_type(8)))  float    v8f;
typedef __attribute__((ext_vector_type(4)))  unsigned int u32x4;
typedef __attribute__((ext_vector_type(8)))  int i32x8;
typedef __attribute__((ext_vector_type(4)))  int i32x4;

#define HIDDEN 1024
#define D_K    64
#define S_LEN  2048
#define NHEADS 16
#define NB     2
#define MROWS  4096   /* NB * S_LEN */

// ---------------------------------------------------------------------------
// Tensor Data Mover: issue an async 2D f16 tile load (rows x cols) from a
// row-major tensor (leading dim = stride_elems) into LDS at lds_off.
// D# layout per CDNA5 ISA 8.3/8.4: group0 = {count/flags, lds_addr,
// global_addr, type=2}; group1 = {data_size=2B, tensor dims, tile dims,
// stride}; groups 2/3 unused (2D tensor).
// ---------------------------------------------------------------------------
__device__ __forceinline__ void tdm_load_2d_f16(
    unsigned lds_off, const _Float16* gptr, unsigned tile_rows,
    unsigned tile_cols, unsigned stride_elems, unsigned tensor_cols,
    unsigned tensor_rows) {
  unsigned long long ga = (unsigned long long)(size_t)gptr;
  u32x4 g0;
  g0[0] = 1u;                                  // count=1 valid, user mode
  g0[1] = lds_off;                             // LDS byte address
  g0[2] = (unsigned)ga;                        // global_addr[31:0]
  g0[3] = (unsigned)(ga >> 32) | (2u << 30);   // global_addr[56:32] | type=2
  i32x8 g1;
  g1[0] = (int)(1u << 16);                     // workgroup_mask=0, data_size=1 (2B)
  g1[1] = (int)((tensor_cols & 0xFFFFu) << 16);                 // dim0[15:0]
  g1[2] = (int)((tensor_cols >> 16) | ((tensor_rows & 0xFFFFu) << 16));
  g1[3] = (int)((tensor_rows >> 16) | (tile_cols << 16));       // tile_dim0
  g1[4] = (int)(tile_rows & 0xFFFFu);          // tile_dim1, tile_dim2=0
  g1[5] = (int)stride_elems;                   // tensor_dim0_stride[31:0]
  g1[6] = 0;
  g1[7] = 0;
  i32x4 z4 = {0, 0, 0, 0};
#if __clang_major__ >= 23
  i32x8 z8 = {0, 0, 0, 0, 0, 0, 0, 0};
  __builtin_amdgcn_tensor_load_to_lds(g0, g1, z4, z4, z8, 0);
#else
  __builtin_amdgcn_tensor_load_to_lds(g0, g1, z4, z4, 0);
#endif
}

// Load a WMMA 16xK-slice fragment (A or B operand) from a row-major f16
// matrix with leading dimension `ld`. CDNA5 16-bit A/B layout (wave32):
// lanes 0-15 hold K = {k0..k0+7, k0+16..k0+23}, lanes 16-31 hold
// K = {k0+8..k0+15, k0+24..k0+31} for row = lane%16.
__device__ __forceinline__ v16h load_frag_rm(const _Float16* base, int ld,
                                             int row, int k0, int lh) {
  const _Float16* p = base + (size_t)row * ld + k0 + lh * 8;
  v8h lo = *(const v8h*)p;
  v8h hi = *(const v8h*)(p + 16);
  v16h r;
#pragma unroll
  for (int i = 0; i < 8; ++i) { r[i] = lo[i]; r[i + 8] = hi[i]; }
  return r;
}

__global__ void cvt_f32_f16(const float* __restrict__ in,
                            _Float16* __restrict__ out, int n) {
  int i = blockIdx.x * blockDim.x + threadIdx.x;
  int stride = gridDim.x * blockDim.x;
  for (; i < n; i += stride) out[i] = (_Float16)in[i];
}

// Y = X @ W^T + bias.  X:[4096][1024] f16, W:[1024][1024] f16 (row n holds k
// contiguously). A/B k-tiles arrive via double-buffered TDM DMA; waves only
// touch LDS + WMMA in the main loop.
// MODE 0: f16 out in [B,H,S,D] head layout.  MODE 1: f32 flat out.
template <int MODE>
__global__ __launch_bounds__(256) void gemm_xwt(
    const _Float16* __restrict__ X, const _Float16* __restrict__ W,
    const float* __restrict__ bias, _Float16* __restrict__ outH,
    float* __restrict__ outF) {
  __shared__ __align__(16) _Float16 As[2][128 * 32];
  __shared__ __align__(16) _Float16 Bs[2][64 * 32];

  const int tid  = threadIdx.x;
  const int wave = tid >> 5;
  const int lane = tid & 31;
  const int lh   = lane >> 4;
  const int l16  = lane & 15;
  const int wm   = wave & 3;   // wave position in M (4)
  const int wn   = wave >> 2;  // wave position in N (2)
  const int m0   = blockIdx.x * 128;
  const int n0   = blockIdx.y * 64;

  v8f acc[2][2] = {};

  // Prime the pipeline: DMA tile kk=0 into buffer 0.
  if (wave == 0) {
    tdm_load_2d_f16((unsigned)(size_t)&As[0][0], X + (size_t)m0 * HIDDEN, 128,
                    32, HIDDEN, HIDDEN, MROWS);
    tdm_load_2d_f16((unsigned)(size_t)&Bs[0][0], W + (size_t)n0 * HIDDEN, 64,
                    32, HIDDEN, HIDDEN, HIDDEN);
  }

  for (int kk = 0; kk < HIDDEN; kk += 32) {
    const int buf = (kk >> 5) & 1;
    if (wave == 0) {
      if (kk + 32 < HIDDEN) {  // issue next tile into alternate buffer first
        tdm_load_2d_f16((unsigned)(size_t)&As[buf ^ 1][0],
                        X + (size_t)m0 * HIDDEN + kk + 32, 128, 32, HIDDEN,
                        HIDDEN, MROWS);
        tdm_load_2d_f16((unsigned)(size_t)&Bs[buf ^ 1][0],
                        W + (size_t)n0 * HIDDEN + kk + 32, 64, 32, HIDDEN,
                        HIDDEN, HIDDEN);
        __builtin_amdgcn_s_wait_tensorcnt(2);  // current tile's 2 DMAs done
      } else {
        __builtin_amdgcn_s_wait_tensorcnt(0);
      }
    }
    __syncthreads();

    v16h a[2], bf[2];
#pragma unroll
    for (int t = 0; t < 2; ++t)
      a[t] = load_frag_rm(&As[buf][0], 32, wm * 32 + t * 16 + l16, 0, lh);
#pragma unroll
    for (int t = 0; t < 2; ++t)
      bf[t] = load_frag_rm(&Bs[buf][0], 32, wn * 32 + t * 16 + l16, 0, lh);

#pragma unroll
    for (int i = 0; i < 2; ++i)
#pragma unroll
      for (int j = 0; j < 2; ++j)
        acc[i][j] = __builtin_amdgcn_wmma_f32_16x16x32_f16(
            false, a[i], false, bf[j], (short)0, acc[i][j], false, false);
    __syncthreads();  // readers done before next DMA overwrites this buffer
  }

  // Epilogue. C/D layout: VGPR r, lane -> row = r + 8*(lane/16), col = lane%16.
#pragma unroll
  for (int i = 0; i < 2; ++i) {
#pragma unroll
    for (int j = 0; j < 2; ++j) {
      int n = n0 + wn * 32 + j * 16 + l16;
      float bv = bias[n];
#pragma unroll
      for (int r = 0; r < 8; ++r) {
        int m = m0 + wm * 32 + i * 16 + r + lh * 8;
        float val = acc[i][j][r] + bv;
        if constexpr (MODE == 0) {
          int bb = m >> 11, s = m & 2047, hh = n >> 6, d = n & 63;
          outH[(((size_t)bb * NHEADS + hh) * S_LEN + s) * D_K + d] =
              (_Float16)val;
        } else {
          outF[(size_t)m * HIDDEN + n] = val;
        }
      }
    }
  }
}

// Flash attention, transposed-score formulation.
// Block = 128 threads (4 waves); each wave owns 16 q rows; 32-key tiles.
// K tile: double-buffered TDM DMA into LDS (shared by all 4 waves).
// V tile: manual LDS transpose (TDM cannot transpose).
__global__ __launch_bounds__(128) void attn_fwd(
    const _Float16* __restrict__ Qh, const _Float16* __restrict__ Kh,
    const _Float16* __restrict__ Vh, const int* __restrict__ mask,
    _Float16* __restrict__ ctx /* [B,S,HIDDEN] concat layout */) {
  __shared__ __align__(16) _Float16 Ks[2][32 * 64];  // K tile, row-major
  __shared__ __align__(16) _Float16 VT[64 * 32];     // V^T tile: [d][key]

  const int tid  = threadIdx.x;
  const int wave = tid >> 5;
  const int lane = tid & 31;
  const int lh   = lane >> 4;
  const int l16  = lane & 15;
  const int qt   = blockIdx.x;
  const int h    = blockIdx.y;
  const int b    = blockIdx.z;

  const size_t headOff = ((size_t)b * NHEADS + h) * (size_t)S_LEN * D_K;
  const _Float16* Q = Qh + headOff;
  const _Float16* K = Kh + headOff;
  const _Float16* V = Vh + headOff;

  const int q0  = qt * 64 + wave * 16;  // this wave's first q row
  const int qme = q0 + l16;             // q whose softmax stats this lane keeps

  // Q as WMMA B operand (q = column), resident in registers for the pass.
  v16h bq_lo = load_frag_rm(Q, D_K, qme, 0, lh);
  v16h bq_hi = load_frag_rm(Q, D_K, qme, 32, lh);

  v8f   acc_o[4] = {};
  float m_run = -3.0e38f, l_run = 0.0f;
  const int* mrow = mask + ((size_t)b * S_LEN + qme) * S_LEN;

  if (wave == 0)  // prime: DMA first K tile
    tdm_load_2d_f16((unsigned)(size_t)&Ks[0][0], K, 32, D_K, D_K, D_K, S_LEN);

  for (int kb = 0; kb < S_LEN; kb += 32) {
    const int buf = (kb >> 5) & 1;

    // Stage V^T (b128 global read, b16 ds scatter transpose).
#pragma unroll
    for (int i = 0; i < 2; ++i) {
      int c = tid + 128 * i;
      int key = c >> 3, ds = (c & 7) * 8;
      v8h t = *(const v8h*)&V[(size_t)(kb + key) * D_K + ds];
#pragma unroll
      for (int j = 0; j < 8; ++j) VT[(ds + j) * 32 + key] = t[j];
    }

    if (wave == 0) {
      if (kb + 32 < S_LEN) {  // issue next K tile, then wait for current one
        tdm_load_2d_f16((unsigned)(size_t)&Ks[buf ^ 1][0],
                        K + (size_t)(kb + 32) * D_K, 32, D_K, D_K, D_K, S_LEN);
        __builtin_amdgcn_s_wait_tensorcnt(1);
      } else {
        __builtin_amdgcn_s_wait_tensorcnt(0);
      }
    }
    __syncthreads();

    // S^T = K_tile(32x64) x Q^T(64x16): rows = keys, cols = q.
    v8f s0 = {}, s1 = {};
    {
      v16h ak = load_frag_rm(&Ks[buf][0], D_K, l16, 0, lh);
      s0 = __builtin_amdgcn_wmma_f32_16x16x32_f16(false, ak, false, bq_lo,
                                                  (short)0, s0, false, false);
      ak = load_frag_rm(&Ks[buf][0], D_K, l16, 32, lh);
      s0 = __builtin_amdgcn_wmma_f32_16x16x32_f16(false, ak, false, bq_hi,
                                                  (short)0, s0, false, false);
      ak = load_frag_rm(&Ks[buf][0], D_K, 16 + l16, 0, lh);
      s1 = __builtin_amdgcn_wmma_f32_16x16x32_f16(false, ak, false, bq_lo,
                                                  (short)0, s1, false, false);
      ak = load_frag_rm(&Ks[buf][0], D_K, 16 + l16, 32, lh);
      s1 = __builtin_amdgcn_wmma_f32_16x16x32_f16(false, ak, false, bq_hi,
                                                  (short)0, s1, false, false);
    }

    // Scale 1/sqrt(64), apply mask; key-axis max = 16 in-lane + 1 shfl_xor.
    float tmax = -3.0e38f;
#pragma unroll
    for (int r = 0; r < 8; ++r) {
      int ki = kb + r + 8 * lh;
      s0[r] = mrow[ki]      ? s0[r] * 0.125f : -1.0e9f;
      s1[r] = mrow[ki + 16] ? s1[r] * 0.125f : -1.0e9f;
      tmax = fmaxf(tmax, fmaxf(s0[r], s1[r]));
    }
    tmax = fmaxf(tmax, __shfl_xor(tmax, 16));

    float m_new = fmaxf(m_run, tmax);
    float corr  = __expf(m_run - m_new);

    // Rescale running context; acc rows are q = r + 8*lh -> broadcast per row.
#pragma unroll
    for (int r = 0; r < 8; ++r) {
      float f = __shfl(corr, r + 8 * lh);
#pragma unroll
      for (int c4 = 0; c4 < 4; ++c4) acc_o[c4][r] *= f;
    }

    // exp(S^T - m): the C layout of S^T *is* the A layout of P (16q x 32key).
    v16h  ap;
    float rs = 0.0f;
#pragma unroll
    for (int r = 0; r < 8; ++r) {
      float e0 = __expf(s0[r] - m_new);
      float e1 = __expf(s1[r] - m_new);
      rs += e0 + e1;
      ap[r]     = (_Float16)e0;
      ap[r + 8] = (_Float16)e1;
    }
    rs += __shfl_xor(rs, 16);
    l_run = l_run * corr + rs;
    m_run = m_new;

    // ctx += P x V, 4 d-chunks of 16; V^T rows in LDS are key-contiguous.
#pragma unroll
    for (int c4 = 0; c4 < 4; ++c4) {
      v16h bv = load_frag_rm(VT, 32, c4 * 16 + l16, 0, lh);
      acc_o[c4] = __builtin_amdgcn_wmma_f32_16x16x32_f16(
          false, ap, false, bv, (short)0, acc_o[c4], false, false);
    }
    __syncthreads();  // readers done before VT/Ks are overwritten
  }

  // Normalize and write concat layout [B, S, H*D] as f16 for the out-proj.
  float linv = 1.0f / l_run;
#pragma unroll
  for (int r = 0; r < 8; ++r) {
    float f = __shfl(linv, r + 8 * lh);
    int qg = q0 + r + 8 * lh;
    _Float16* orow = ctx + ((size_t)b * S_LEN + qg) * HIDDEN + h * D_K;
#pragma unroll
    for (int c4 = 0; c4 < 4; ++c4)
      orow[c4 * 16 + l16] = (_Float16)(acc_o[c4][r] * f);
  }
}

extern "C" void kernel_launch(void* const* d_in, const int* in_sizes, int n_in,
                              void* d_out, int out_size, void* d_ws,
                              size_t ws_size, hipStream_t stream) {
  const float* q    = (const float*)d_in[0];
  const float* k    = (const float*)d_in[1];
  const float* v    = (const float*)d_in[2];
  const int*   mask = (const int*)d_in[3];
  const float* Wq   = (const float*)d_in[4];
  const float* bq   = (const float*)d_in[5];
  const float* Wk   = (const float*)d_in[6];
  const float* bk   = (const float*)d_in[7];
  const float* Wv   = (const float*)d_in[8];
  const float* bv   = (const float*)d_in[9];
  const float* Wo   = (const float*)d_in[10];
  const float* bo   = (const float*)d_in[11];

  // Workspace layout (f16 elements); total = 7*4M + 4*1M elems = 64 MB.
  const size_t NE_X = (size_t)MROWS * HIDDEN;   // 4194304
  const size_t NE_W = (size_t)HIDDEN * HIDDEN;  // 1048576
  _Float16* wsp = (_Float16*)d_ws;
  _Float16* qf  = wsp;
  _Float16* kf  = qf + NE_X;
  _Float16* vf  = kf + NE_X;
  _Float16* Wqf = vf + NE_X;
  _Float16* Wkf = Wqf + NE_W;
  _Float16* Wvf = Wkf + NE_W;
  _Float16* Wof = Wvf + NE_W;
  _Float16* Qh  = Wof + NE_W;
  _Float16* Kh  = Qh + NE_X;
  _Float16* Vh  = Kh + NE_X;
  _Float16* ctx = Vh + NE_X;

  // 1) fp32 -> f16 conversions
  cvt_f32_f16<<<2048, 256, 0, stream>>>(q, qf, (int)NE_X);
  cvt_f32_f16<<<2048, 256, 0, stream>>>(k, kf, (int)NE_X);
  cvt_f32_f16<<<2048, 256, 0, stream>>>(v, vf, (int)NE_X);
  cvt_f32_f16<<<1024, 256, 0, stream>>>(Wq, Wqf, (int)NE_W);
  cvt_f32_f16<<<1024, 256, 0, stream>>>(Wk, Wkf, (int)NE_W);
  cvt_f32_f16<<<1024, 256, 0, stream>>>(Wv, Wvf, (int)NE_W);
  cvt_f32_f16<<<1024, 256, 0, stream>>>(Wo, Wof, (int)NE_W);

  // 2) QKV projections straight into [B,H,S,D] head layout (f16)
  dim3 ggrid(MROWS / 128, HIDDEN / 64);
  gemm_xwt<0><<<ggrid, 256, 0, stream>>>(qf, Wqf, bq, Qh, nullptr);
  gemm_xwt<0><<<ggrid, 256, 0, stream>>>(kf, Wkf, bk, Kh, nullptr);
  gemm_xwt<0><<<ggrid, 256, 0, stream>>>(vf, Wvf, bv, Vh, nullptr);

  // 3) flash attention -> ctx in [B,S,HIDDEN] concat layout (f16)
  attn_fwd<<<dim3(S_LEN / 64, NHEADS, NB), 128, 0, stream>>>(Qh, Kh, Vh, mask,
                                                             ctx);

  // 4) output projection, fp32 result
  gemm_xwt<1><<<ggrid, 256, 0, stream>>>(ctx, Wof, bo, nullptr, (float*)d_out);
}
